// InrAttn_17738214933127
// MI455X (gfx1250) — matrix operand order repaired and checked by reading, not compiled, not run on hardware
//
#include <hip/hip_runtime.h>

// CDNA5 (gfx1250) fused "InrAttn": out = softmax(QK^T/8 + softmax(masked str)) @ V
// Flash-style single pass over k, WMMA f16 16x16x32 with f32 accumulation.
// All selects are input-selects (v_cndmask) so EXEC stays all-1s around WMMA.

typedef __attribute__((ext_vector_type(16))) _Float16 v16h;
typedef __attribute__((ext_vector_type(8)))  _Float16 v8h;
typedef __attribute__((ext_vector_type(8)))  float    v8f;
typedef __attribute__((ext_vector_type(4)))  float    f32x4;
typedef __attribute__((ext_vector_type(4)))  int      i32x4;

constexpr int Bc = 2, Hc = 8, Sc = 2048, Dc = 64;
constexpr int QT = 32;   // q rows per workgroup
constexpr int KT = 32;   // k columns per chunk (one WMMA K-dim for PV)

__global__ __launch_bounds__(256)
void inr_attn_kernel(const float* __restrict__ Q,
                     const float* __restrict__ K,
                     const float* __restrict__ V,
                     const float* __restrict__ STR,
                     const int*   __restrict__ MASK,
                     float* __restrict__ O)
{
    __shared__ __align__(64) _Float16 Klds[KT][Dc];     // K chunk, row-major (k, d), f16
    __shared__ __align__(64) _Float16 Vtlds[Dc][KT];    // V chunk transposed (d, k), f16
    __shared__ __align__(64) _Float16 Plds[8][16][KT];  // per-wave P staging (m, k)
    __shared__ float s1inv[QT];                         // 1 / sum(exp(str_masked)) per row

    const int tid  = threadIdx.x;
    const int lane = tid & 31;
    const int wid  = tid >> 5;
    const int lo   = lane & 15;     // n / m-low index inside fragments
    const int hi   = lane >> 4;     // half-wave select
    const int qb   = wid >> 2;      // q block (0..1): rows qb*16 .. qb*16+15
    const int nb   = wid & 3;       // d block (0..3): cols nb*16 .. nb*16+15

    const int qt = blockIdx.x * QT;
    const int bh = blockIdx.y;
    const int b  = bh / Hc;

    // ---------------- Pass A: row sums of exp(masked str) ----------------
    // str ~ N(0,1); masked entries -> -1e9 -> exp underflows to +0.
    // Input-select (cndmask), exp always executed: no EXEC manipulation.
    #pragma unroll
    for (int rr = 0; rr < QT / 8; ++rr) {     // 4 rows per wave
        const int qloc = wid * (QT / 8) + rr;
        const int q    = qt + qloc;
        const float* srow = STR  + ((size_t)bh * Sc + q) * (size_t)Sc;
        const int*   mrow = MASK + ((size_t)b  * Sc + q) * (size_t)Sc;
        float sum = 0.f;
        for (int it = 0; it < Sc / 128; ++it) {        // exactly 16 iters, uniform
            const int k0 = lane * 4 + it * 128;
            f32x4 x = *(const f32x4*)(srow + k0);
            i32x4 m = *(const i32x4*)(mrow + k0);
            __builtin_prefetch(srow + k0 + 1024, 0, 0);  // global_prefetch_b8
            #pragma unroll
            for (int j = 0; j < 4; ++j) {
                const float xm = (m[j] != 0) ? x[j] : -1e9f;
                sum += __expf(xm);
            }
        }
        #pragma unroll
        for (int off = 16; off >= 1; off >>= 1)
            sum += __shfl_xor(sum, off, 32);
        if (lane == 0) s1inv[qloc] = 1.0f / fmaxf(sum, 1e-30f);
    }
    __syncthreads();

    // ---------------- Q fragments (A-layout), scale 1/8 folded in --------
    const int qrow = qt + qb * 16 + lo;                     // A-layout: m = lane&15
    const float* qptr = Q + ((size_t)bh * Sc + qrow) * (size_t)Dc;
    v16h qf0, qf1;
    #pragma unroll
    for (int e = 0; e < 16; ++e) {
        const int d = e + 16 * hi;                          // A-layout k index
        qf0[e] = (_Float16)(0.125f * qptr[d]);
        qf1[e] = (_Float16)(0.125f * qptr[d + 32]);
    }

    // Per-row bias normalizers and str/mask row pointers (C-layout rows)
    float binv[8];
    const float* srowp[8];
    const int*   mrowp[8];
    #pragma unroll
    for (int r = 0; r < 8; ++r) {
        const int qloc = qb * 16 + r + 8 * hi;              // C-layout: m = r + 8*hi
        binv[r]  = s1inv[qloc];
        srowp[r] = STR  + ((size_t)bh * Sc + (qt + qloc)) * (size_t)Sc;
        mrowp[r] = MASK + ((size_t)b  * Sc + (qt + qloc)) * (size_t)Sc;
    }

    v8f   oacc = {};                          // PV accumulator (C-layout)
    float lsum[8] = {0, 0, 0, 0, 0, 0, 0, 0}; // running sum(exp(scores)) per row

    // ---------------- Pass B: flash loop over k chunks -------------------
    for (int kc = 0; kc < Sc / KT; ++kc) {
        const int k0 = kc * KT;

        // Cooperative stage: K chunk (f32->f16) and V chunk transposed.
        // Compile-time trip count: no exec-masked loop.
        const float* kbase = K + ((size_t)bh * Sc + k0) * (size_t)Dc;
        const float* vbase = V + ((size_t)bh * Sc + k0) * (size_t)Dc;
        #pragma unroll
        for (int j = 0; j < (KT * Dc) / 256; ++j) {   // 8 iterations
            const int i  = tid + 256 * j;
            const int kk = i >> 6;           // /Dc
            const int d  = i & (Dc - 1);
            Klds[kk][d]  = (_Float16)kbase[i];
            Vtlds[d][kk] = (_Float16)vbase[i];
        }
        __syncthreads();

        #pragma unroll
        for (int tile = 0; tile < 2; ++tile) {
            // B-fragment of K^T: n = kcol, k = d
            const int kcol = tile * 16 + lo;
            v16h kf0 = *(const v16h*)(&Klds[kcol][16 * hi]);        // d 0..31
            v16h kf1 = *(const v16h*)(&Klds[kcol][16 * hi + 32]);   // d 32..63
            v8f sc = {};
            sc = __builtin_amdgcn_wmma_f32_16x16x32_f16(false, qf0, false, kf0,
                                                        (short)0, sc, false, false);
            sc = __builtin_amdgcn_wmma_f32_16x16x32_f16(false, qf1, false, kf1,
                                                        (short)0, sc, false, false);
            const int kg = k0 + tile * 16 + lo;  // global k for this lane's column
            #pragma unroll
            for (int r = 0; r < 8; ++r) {
                const float sv = srowp[r][kg];
                const int   mv = mrowp[r][kg];
                const float xm = (mv != 0) ? sv : -1e9f;   // input-select, no EXEC
                const float bias = __expf(xm) * binv[r];
                const float p    = __expf(sc[r] + bias);
                lsum[r] += p;
                Plds[wid][r + 8 * hi][tile * 16 + lo] = (_Float16)p;  // C -> LDS
            }
        }
        // Same-wave LDS ordering (DS ops are in-order per wave); pin compiler.
        asm volatile("s_wait_dscnt 0x0" ::: "memory");

        // Re-read P in A-fragment layout: m = lo, k = e + 16*hi
        v8h p0 = *(const v8h*)(&Plds[wid][lo][16 * hi]);
        v8h p1 = *(const v8h*)(&Plds[wid][lo][16 * hi + 8]);
        v16h pf;
        #pragma unroll
        for (int e = 0; e < 8; ++e) { pf[e] = p0[e]; pf[8 + e] = p1[e]; }

        // B-fragment of V: n = d column, k = chunk row
        v16h vf = *(const v16h*)(&Vtlds[nb * 16 + lo][16 * hi]);
        oacc = __builtin_amdgcn_wmma_f32_16x16x32_f16(false, pf, false, vf,
                                                      (short)0, oacc, false, false);
        __syncthreads();   // protect Klds/Vtlds before next chunk's staging
    }

    // ---------------- Normalize and store --------------------------------
    #pragma unroll
    for (int r = 0; r < 8; ++r) {
        float l = lsum[r];
        #pragma unroll
        for (int off = 8; off >= 1; off >>= 1)   // reduce within 16-lane half
            l += __shfl_xor(l, off, 32);
        const int q = qt + qb * 16 + r + 8 * hi;
        O[((size_t)bh * Sc + q) * (size_t)Dc + nb * 16 + lo] = oacc[r] / l;
    }
}

extern "C" void kernel_launch(void* const* d_in, const int* in_sizes, int n_in,
                              void* d_out, int out_size, void* d_ws, size_t ws_size,
                              hipStream_t stream) {
    const float* Q    = (const float*)d_in[0];
    const float* K    = (const float*)d_in[1];
    const float* V    = (const float*)d_in[2];
    const float* STRm = (const float*)d_in[3];
    const int*   MSK  = (const int*)d_in[4];
    float* O = (float*)d_out;

    dim3 grid(Sc / QT, Bc * Hc);   // (64, 16) workgroups
    inr_attn_kernel<<<grid, 256, 0, stream>>>(Q, K, V, STRm, MSK, O);

    (void)in_sizes; (void)n_in; (void)out_size; (void)d_ws; (void)ws_size;
}
